// DeformationNetworkGraphConvolutionalFull_29368986370496
// MI455X (gfx1250) — matrix operand order repaired and checked by reading, compiled, not run on hardware
//
#include <hip/hip_runtime.h>

// ---------------------------------------------------------------------------
// CDNA5 (gfx1250): bf16 WMMA GEMMs (f32 accum), fragment-major LDS staging,
// software-pipelined fragment loads, atomic f32 edge scatter.
// Wave32, 8 waves/block, 128x64 C-tile per block.
// ---------------------------------------------------------------------------

typedef __attribute__((ext_vector_type(16))) __bf16 v16bf;
typedef __attribute__((ext_vector_type(8)))  float  v8f;

#define MBLK   128          // C-tile rows per block (8 waves x 16)
#define NBLK   64           // C-tile cols per block (4 x 16 per wave)
#define KSTEP  64           // two K=32 WMMA steps per staging phase
#define FPITCH 12           // u32 words per 8-word fragment (48B, 16B-aligned)

__device__ __forceinline__ unsigned int pack_bf16(float a, float b) {
  // round-to-nearest-even f32 -> bf16; low = even-k, high = odd-k
  unsigned int ua = __float_as_uint(a);
  unsigned int ub = __float_as_uint(b);
  ua = (ua + 0x7FFFu + ((ua >> 16) & 1u)) >> 16;
  ub = (ub + 0x7FFFu + ((ub >> 16) & 1u)) >> 16;
  return ua | (ub << 16);
}

// C[M,N] = act( A[M,K] @ W[K,N] + bias[N] + extraA[M,extraK] @ extraW[extraK,N] )
// K must be a multiple of 64.  gridDim.y covers N in NBLK chunks.
__global__ __launch_bounds__(256)
void gemm_bf16_wmma(const float* __restrict__ A, const float* __restrict__ W,
                    const float* __restrict__ bias,
                    const float* __restrict__ extraA,
                    const float* __restrict__ extraW, int extraK,
                    float* __restrict__ C, int M, int K, int N, int do_relu) {
  // A fragments: [row 0..127][ks 0..1][hlf 0..1] -> 8 words @ pitch FPITCH
  __shared__ unsigned int Asmem[MBLK * 2 * 2 * FPITCH];      // 24 KB
  // B fragments: [nt 0..3][ks 0..1][lane 0..31] -> 8 words @ pitch FPITCH
  __shared__ unsigned int Bsmem[4 * 2 * 32 * FPITCH];        // 12 KB

  const int tid  = threadIdx.x;
  const int wave = tid >> 5;
  const int lane = tid & 31;
  const int hlf  = lane >> 4;      // 0: lanes 0-15, 1: lanes 16-31
  const int l16  = lane & 15;
  const int m0   = blockIdx.x * MBLK;
  const int n0   = blockIdx.y * NBLK;
  const bool full_tile = (m0 + MBLK) <= M;   // uniform: 160/161 blocks

  v8f acc[4];
  const v8f vzero = {0.f, 0.f, 0.f, 0.f, 0.f, 0.f, 0.f, 0.f};
#pragma unroll
  for (int t = 0; t < 4; ++t) acc[t] = vzero;

  union Frag { uint4 q[2]; v16bf v; };

  for (int k0 = 0; k0 < K; k0 += KSTEP) {
    // ---- stage A: float4 -> two packed bf16 words at adjacent frag slots ---
    // quad q -> row = q/16, fq = q%16 covers k = 4*fq .. 4*fq+3 (kp = 2fq,2fq+1)
    if (full_tile) {
#pragma unroll
      for (int q = tid; q < MBLK * (KSTEP / 4); q += 256) {
        int row = q >> 4;
        int fq  = q & 15;
        float4 av = *(const float4*)(A + (size_t)(m0 + row) * K + k0 + 4 * fq);
        int kp  = 2 * fq;
        int ks  = kp >> 4;
        int kpl = kp & 15;
        int h   = (kpl >> 2) & 1;
        int r   = (kpl & 3) + ((kpl >> 3) << 2);
        int d   = ((row * 2 + ks) * 2 + h) * FPITCH + r;   // d even -> b64 store
        Asmem[d]     = pack_bf16(av.x, av.y);
        Asmem[d + 1] = pack_bf16(av.z, av.w);
      }
    } else {
#pragma unroll
      for (int q = tid; q < MBLK * (KSTEP / 4); q += 256) {
        int row = q >> 4;
        int fq  = q & 15;
        int gm  = m0 + row;
        float4 av = {0.f, 0.f, 0.f, 0.f};
        if (gm < M) av = *(const float4*)(A + (size_t)gm * K + k0 + 4 * fq);
        int kp  = 2 * fq;
        int ks  = kp >> 4;
        int kpl = kp & 15;
        int h   = (kpl >> 2) & 1;
        int r   = (kpl & 3) + ((kpl >> 3) << 2);
        int d   = ((row * 2 + ks) * 2 + h) * FPITCH + r;
        Asmem[d]     = pack_bf16(av.x, av.y);
        Asmem[d + 1] = pack_bf16(av.z, av.w);
      }
    }
    // ---- stage B: pack (k,k+1) column pairs into fragment-major slots ------
    // quad q -> kp = q/16 (0..31), n4 = 4*(q%16)
#pragma unroll
    for (int q = tid; q < (KSTEP / 2) * (NBLK / 4); q += 256) {
      int kp = q >> 4;
      int n4 = (q & 15) * 4;
      int gk = k0 + 2 * kp;
      float4 be = *(const float4*)(W + (size_t)gk * N + n0 + n4);
      float4 bo = *(const float4*)(W + (size_t)(gk + 1) * N + n0 + n4);
      int ks  = kp >> 4;
      int kpl = kp & 15;
      int h   = (kpl >> 2) & 1;
      int r   = (kpl & 3) + ((kpl >> 3) << 2);
      int nt  = n4 >> 4;
      int l   = n4 & 15;
      int base = ((nt * 2 + ks) * 32 + h * 16 + l) * FPITCH + r;
      Bsmem[base + 0 * FPITCH] = pack_bf16(be.x, bo.x);
      Bsmem[base + 1 * FPITCH] = pack_bf16(be.y, bo.y);
      Bsmem[base + 2 * FPITCH] = pack_bf16(be.z, bo.z);
      Bsmem[base + 3 * FPITCH] = pack_bf16(be.w, bo.w);
    }
    // ---- prefetch next K tile (global_prefetch_b8) -------------------------
    if (k0 + KSTEP < K) {
      int pr = m0 + (tid & 127);
      if (pr < M) __builtin_prefetch(A + (size_t)pr * K + k0 + KSTEP, 0, 0);
      int pk = k0 + KSTEP + (tid & 63);
      if (pk < K) __builtin_prefetch(W + (size_t)pk * N + n0, 0, 0);
    }
    __syncthreads();

    // ---- compute: 8 WMMAs, software-pipelined fragment loads ---------------
    const int arow = wave * 16 + l16;
    Frag af[2];
#pragma unroll
    for (int ks = 0; ks < 2; ++ks) {
      const uint4* ap = (const uint4*)&Asmem[((arow * 2 + ks) * 2 + hlf) * FPITCH];
      af[ks].q[0] = ap[0];
      af[ks].q[1] = ap[1];
    }
    Frag bf[2];
    {
      const uint4* bp = (const uint4*)&Bsmem[(0 * 32 + lane) * FPITCH];
      bf[0].q[0] = bp[0];
      bf[0].q[1] = bp[1];
    }
#pragma unroll
    for (int s = 0; s < 8; ++s) {
      // step s -> (nt = s>>1, ks = s&1)  [matches Bsmem (nt*2+ks) ordering]
      int nt = s >> 1;
      int ks = s & 1;
      if (s < 7) {
        const uint4* bp = (const uint4*)&Bsmem[((s + 1) * 32 + lane) * FPITCH];
        bf[(s + 1) & 1].q[0] = bp[0];
        bf[(s + 1) & 1].q[1] = bp[1];
      }
      acc[nt] = __builtin_amdgcn_wmma_f32_16x16x32_bf16(
          false, af[ks].v, false, bf[s & 1].v, (short)0, acc[nt], false, false);
    }
    __syncthreads();
  }

  // ---- epilogue: bias + concat-tail + relu (ISA C/D layout) ----------------
#pragma unroll
  for (int nt = 0; nt < 4; ++nt) {
#pragma unroll
    for (int r = 0; r < 8; ++r) {
      int m = m0 + wave * 16 + r + hlf * 8;
      int n = n0 + nt * 16 + l16;
      if (m < M) {
        float c = acc[nt][r] + bias[n];
        for (int t = 0; t < extraK; ++t)
          c += extraA[(size_t)m * extraK + t] * extraW[(size_t)t * N + n];
        if (do_relu) c = fmaxf(c, 0.f);
        C[(size_t)m * N + n] = c;
      }
    }
  }
}

// out[i] += nbr[j]; out[j] += nbr[i]  (per edge, per feature) via f32 atomics
__global__ __launch_bounds__(256)
void scatter_edges(float* __restrict__ out, const float* __restrict__ nbr,
                   const int* __restrict__ edges, int E, int H) {
  int gid = blockIdx.x * 256 + threadIdx.x;
  if (gid >= E * H) return;
  int e, f;
  if (H == 256) { e = gid >> 8; f = gid & 255; }
  else          { e = gid / H;  f = gid - e * H; }
  int i = edges[2 * e + 0];
  int j = edges[2 * e + 1];
  atomicAdd(&out[(size_t)i * H + f], nbr[(size_t)j * H + f]);
  atomicAdd(&out[(size_t)j * H + f], nbr[(size_t)i * H + f]);
}

__global__ __launch_bounds__(256)
void relu_inplace(float* __restrict__ x, int n) {
  int gid = blockIdx.x * 256 + threadIdx.x;
  if (gid < n) x[gid] = fmaxf(x[gid], 0.f);
}

// tiny output heads: thread per (m,n); act: 0=none, 1=relu, 2=sigmoid
__global__ __launch_bounds__(256)
void small_gemm(const float* __restrict__ A, const float* __restrict__ W,
                const float* __restrict__ bias, float* __restrict__ C,
                int M, int K, int N, int act) {
  int gid = blockIdx.x * 256 + threadIdx.x;
  if (gid >= M * N) return;
  int m = gid / N;
  int n = gid - m * N;
  float acc = bias[n];
  const float* a = A + (size_t)m * K;
  for (int k = 0; k < K; ++k) acc = fmaf(a[k], W[(size_t)k * N + n], acc);
  if (act == 1) acc = fmaxf(acc, 0.f);
  else if (act == 2) acc = 1.f / (1.f + __expf(-acc));
  C[(size_t)m * N + n] = acc;
}

extern "C" void kernel_launch(void* const* d_in, const int* in_sizes, int n_in,
                              void* d_out, int out_size, void* d_ws, size_t ws_size,
                              hipStream_t stream) {
  const float* feats = (const float*)d_in[0];
  const float* verts = (const float*)d_in[1];
  const int*   edges = (const int*)d_in[2];
  const float* Wb    = (const float*)d_in[3];
  const float* bb    = (const float*)d_in[4];
  const float* W0_0  = (const float*)d_in[5];
  const float* b0_0  = (const float*)d_in[6];
  const float* W1_0  = (const float*)d_in[7];
  const float* b1_0  = (const float*)d_in[8];
  const float* gW0   = (const float*)d_in[9];
  const float* gb0   = (const float*)d_in[10];
  const float* gW1   = (const float*)d_in[11];
  const float* gb1   = (const float*)d_in[12];
  const float* Wo    = (const float*)d_in[13];
  const float* bo    = (const float*)d_in[14];
  const float* A1    = (const float*)d_in[15];
  const float* a1    = (const float*)d_in[16];
  const float* A2    = (const float*)d_in[17];
  const float* a2    = (const float*)d_in[18];
  const float* A3    = (const float*)d_in[19];
  const float* a3    = (const float*)d_in[20];
  const float* A4    = (const float*)d_in[21];
  const float* a4    = (const float*)d_in[22];

  const int V = in_sizes[1] / 3;    // 20496
  const int E = in_sizes[2] / 2;    // 61440
  const int F = in_sizes[0] / V;    // 3840
  const int H = in_sizes[4];        // 256

  float* h   = (float*)d_ws;                 // V*H
  float* out = h   + (size_t)V * H;          // V*H
  float* nbr = out + (size_t)V * H;          // V*H

  float* delta_v = (float*)d_out;            // V*3
  float* conf    = delta_v + (size_t)V * 3;  // V*1

  const dim3 blk(256);
  const dim3 gH((V + MBLK - 1) / MBLK, H / NBLK);        // N = 256
  const dim3 g64((V + MBLK - 1) / MBLK, 64 / NBLK);      // N = 64
  const int relu_blocks    = (V * H + 255) / 256;
  const int scatter_blocks = (E * H + 255) / 256;

  // 1) h = relu(feats @ Wb + bb)      [V,3840]x[3840,256]
  gemm_bf16_wmma<<<gH, blk, 0, stream>>>(feats, Wb, bb, nullptr, nullptr, 0,
                                         h, V, F, H, 1);

  float* cur = h;
  float* buf = out;

  // 2) conv0: concat([h,verts]) handled as K=256 WMMA + 3-row fp32 tail
  gemm_bf16_wmma<<<gH, blk, 0, stream>>>(cur, W0_0, b0_0, verts,
                                         W0_0 + (size_t)H * H, 3,
                                         buf, V, H, H, 0);
  gemm_bf16_wmma<<<gH, blk, 0, stream>>>(cur, W1_0, b1_0, verts,
                                         W1_0 + (size_t)H * H, 3,
                                         nbr, V, H, H, 0);
  scatter_edges<<<scatter_blocks, blk, 0, stream>>>(buf, nbr, edges, E, H);
  relu_inplace<<<relu_blocks, blk, 0, stream>>>(buf, V * H);
  { float* t = cur; cur = buf; buf = t; }

  // 3) 9 graph-conv layers
  for (int k = 0; k < 9; ++k) {
    gemm_bf16_wmma<<<gH, blk, 0, stream>>>(cur, gW0 + (size_t)k * H * H,
                                           gb0 + (size_t)k * H,
                                           nullptr, nullptr, 0,
                                           buf, V, H, H, 0);
    gemm_bf16_wmma<<<gH, blk, 0, stream>>>(cur, gW1 + (size_t)k * H * H,
                                           gb1 + (size_t)k * H,
                                           nullptr, nullptr, 0,
                                           nbr, V, H, H, 0);
    scatter_edges<<<scatter_blocks, blk, 0, stream>>>(buf, nbr, edges, E, H);
    relu_inplace<<<relu_blocks, blk, 0, stream>>>(buf, V * H);
    float* t = cur; cur = buf; buf = t;
  }

  // 4) delta_v = cur @ Wo + bo   [V,256]x[256,3]
  small_gemm<<<(V * 3 + 255) / 256, blk, 0, stream>>>(cur, Wo, bo, delta_v,
                                                      V, H, 3, 0);

  // 5) MLP head: z1 = relu(cur@A1+a1); z2 = relu(z1@A2+a2)
  gemm_bf16_wmma<<<gH, blk, 0, stream>>>(cur, A1, a1, nullptr, nullptr, 0,
                                         buf, V, H, H, 1);
  gemm_bf16_wmma<<<g64, blk, 0, stream>>>(buf, A2, a2, nullptr, nullptr, 0,
                                          nbr, V, H, 64, 1);
  // z3 = relu(z2@A3+a3)  -> reuse `cur` (no longer needed)
  small_gemm<<<(V * 32 + 255) / 256, blk, 0, stream>>>(nbr, A3, a3, cur,
                                                       V, 64, 32, 1);
  // conf = sigmoid(z3@A4+a4)
  small_gemm<<<(V * 1 + 255) / 256, blk, 0, stream>>>(cur, A4, a4, conf,
                                                      V, 32, 1, 2);
}
